// BinaryLinear_49168785605335
// MI455X (gfx1250) — matrix operand order repaired and checked by reading, not compiled, not run on hardware
//
#include <hip/hip_runtime.h>
#include <hip/hip_bf16.h>

typedef __attribute__((ext_vector_type(16))) __bf16 v16bf;
typedef __attribute__((ext_vector_type(8)))  float  v8f;

#define M_ROWS 8192
#define K_DIM  4096
#define N_DIM  4096

constexpr int BM  = 128;          // output tile rows per block
constexpr int BN  = 128;          // output tile cols per block
constexpr int BK  = 32;           // k-step (matches 16x16x32 bf16 WMMA)
constexpr int LDT = BK + 8;       // padded LDS row stride (40 bf16 = 80 B -> conflict-free frag reads)
constexpr int NKT = K_DIM / BK;   // 128 k-tiles

union FragU { uint4 u[2]; v16bf v; };

__device__ __forceinline__ unsigned pack2_bf16(float a, float b) {
  union { __bf16 h[2]; unsigned u; } r;
  r.h[0] = (__bf16)a;
  r.h[1] = (__bf16)b;
  return r.u;
}

// ---------------------------------------------------------------------------
// Prepass 1: x (fp32) -> bf16, 8 elements / thread
// ---------------------------------------------------------------------------
__global__ __launch_bounds__(256)
void cvt_x_bf16(const float* __restrict__ src, __bf16* __restrict__ dst) {
  size_t gid  = (size_t)blockIdx.x * 256 + threadIdx.x;
  size_t i8   = gid * 8;
  float4 a = *(const float4*)(src + i8);
  float4 b = *(const float4*)(src + i8 + 4);
  uint4 p;
  p.x = pack2_bf16(a.x, a.y);
  p.y = pack2_bf16(a.z, a.w);
  p.z = pack2_bf16(b.x, b.y);
  p.w = pack2_bf16(b.z, b.w);
  *(uint4*)(dst + i8) = p;
}

// ---------------------------------------------------------------------------
// Prepass 2: W (fp32) -> sign(W)*alpha[row] in bf16, 8 elements / thread
// ---------------------------------------------------------------------------
__global__ __launch_bounds__(256)
void cvt_w_bf16(const float* __restrict__ W, const float* __restrict__ alpha,
                __bf16* __restrict__ dst) {
  size_t gid  = (size_t)blockIdx.x * 256 + threadIdx.x;
  size_t i8   = gid * 8;
  int row = (int)(i8 >> 12);        // K_DIM = 4096; all 8 elems share a row
  float al = alpha[row];
  float4 a = *(const float4*)(W + i8);
  float4 b = *(const float4*)(W + i8 + 4);
  uint4 p;
  p.x = pack2_bf16(__builtin_copysignf(al, a.x), __builtin_copysignf(al, a.y));
  p.y = pack2_bf16(__builtin_copysignf(al, a.z), __builtin_copysignf(al, a.w));
  p.z = pack2_bf16(__builtin_copysignf(al, b.x), __builtin_copysignf(al, b.y));
  p.w = pack2_bf16(__builtin_copysignf(al, b.z), __builtin_copysignf(al, b.w));
  *(uint4*)(dst + i8) = p;
}

// ---------------------------------------------------------------------------
// Fast GEMM: bf16 operands from workspace, async global->LDS staging (ASYNCcnt)
// ---------------------------------------------------------------------------
__global__ __launch_bounds__(256)
void binlin_wmma_bf16_async(const __bf16* __restrict__ xb,
                            const __bf16* __restrict__ wb,
                            const float* __restrict__ bias,
                            float* __restrict__ out)
{
  __shared__ __align__(16) __bf16 As[2][BM * LDT];
  __shared__ __align__(16) __bf16 Bs[2][BM * LDT];

  const int tid  = threadIdx.x;
  const int lane = tid & 31;
  const int wid  = tid >> 5;
  const int waveM = wid & 3;
  const int waveN = wid >> 2;

  const int m0 = blockIdx.y * BM;
  const int n0 = blockIdx.x * BN;

  // Per-thread async-fill slots: tile = 128 rows x 32 bf16 = 512 x 16B chunks,
  // 256 threads -> 2 chunks each per matrix.  row = idx>>2, chunk = idx&3.
  unsigned long long gA[2], gB[2];
  unsigned lA[2], lB[2];
  const unsigned ldsA0 = (unsigned)(size_t)(void*)&As[0][0];
  const unsigned ldsB0 = (unsigned)(size_t)(void*)&Bs[0][0];
  #pragma unroll
  for (int j = 0; j < 2; ++j) {
    int idx = tid + 256 * j;        // 0..511
    int row = idx >> 2;             // 0..127
    int ch  = idx & 3;              // 16B chunk within the 64B row
    gA[j] = (unsigned long long)(size_t)(xb + (size_t)(m0 + row) * K_DIM + ch * 8);
    gB[j] = (unsigned long long)(size_t)(wb + (size_t)(n0 + row) * K_DIM + ch * 8);
    lA[j] = ldsA0 + (unsigned)(row * LDT + ch * 8) * 2;
    lB[j] = ldsB0 + (unsigned)(row * LDT + ch * 8) * 2;
  }

  auto fill_async = [&](int kt, int s) {
    const unsigned long long go = (unsigned long long)kt * (BK * 2); // 64 B per k-tile
    const unsigned so = (unsigned)s * (BM * LDT * 2);
    #pragma unroll
    for (int j = 0; j < 2; ++j) {
      unsigned long long ga = gA[j] + go;
      unsigned long long gb = gB[j] + go;
      unsigned la = lA[j] + so;
      unsigned lb = lB[j] + so;
      asm volatile("global_load_async_to_lds_b128 %0, %1, off"
                   :: "v"(la), "v"(ga) : "memory");
      asm volatile("global_load_async_to_lds_b128 %0, %1, off"
                   :: "v"(lb), "v"(gb) : "memory");
    }
  };

  v8f acc[2][4];
  const v8f vzero = {0.f, 0.f, 0.f, 0.f, 0.f, 0.f, 0.f, 0.f};
  #pragma unroll
  for (int mi = 0; mi < 2; ++mi)
    #pragma unroll
    for (int ni = 0; ni < 4; ++ni)
      acc[mi][ni] = vzero;

  fill_async(0, 0);
  asm volatile("s_wait_asynccnt 0x0" ::: "memory");
  __syncthreads();

  const int half = lane >> 4;
  const int l15  = lane & 15;

  for (int kt = 0; kt < NKT; ++kt) {
    const int s = kt & 1;

    // kick off the next stage copy before touching this stage
    if (kt + 1 < NKT) fill_async(kt + 1, s ^ 1);

    v16bf af[2];
    #pragma unroll
    for (int mi = 0; mi < 2; ++mi) {
      int row = waveM * 32 + mi * 16 + l15;
      const char* p = (const char*)&As[s][row * LDT];
      int o = half ? 16 : 0;
      FragU f;
      f.u[0] = *(const uint4*)(p + o);
      f.u[1] = *(const uint4*)(p + o + 32);
      af[mi] = f.v;
    }
    v16bf bfr[4];
    #pragma unroll
    for (int ni = 0; ni < 4; ++ni) {
      int col = waveN * 64 + ni * 16 + l15;
      const char* p = (const char*)&Bs[s][col * LDT];
      int o = half ? 16 : 0;
      FragU f;
      f.u[0] = *(const uint4*)(p + o);
      f.u[1] = *(const uint4*)(p + o + 32);
      bfr[ni] = f.v;
    }

    #pragma unroll
    for (int mi = 0; mi < 2; ++mi)
      #pragma unroll
      for (int ni = 0; ni < 4; ++ni)
        acc[mi][ni] = __builtin_amdgcn_wmma_f32_16x16x32_bf16(
            false, af[mi], false, bfr[ni], (short)0, acc[mi][ni], false, false);

    // next-stage async copies must land before the barrier releases readers
    asm volatile("s_wait_asynccnt 0x0" ::: "memory");
    __syncthreads();
  }

  #pragma unroll
  for (int mi = 0; mi < 2; ++mi) {
    int mbase = m0 + waveM * 32 + mi * 16 + half * 8;
    #pragma unroll
    for (int ni = 0; ni < 4; ++ni) {
      int n = n0 + waveN * 64 + ni * 16 + l15;
      float bv = bias[n];
      #pragma unroll
      for (int r = 0; r < 8; ++r)
        out[(size_t)(mbase + r) * N_DIM + n] = acc[mi][ni][r] + bv;
    }
  }
}

// ---------------------------------------------------------------------------
// Fallback GEMM (in-kernel fp32->bf16 conversion; used if ws is too small)
// ---------------------------------------------------------------------------
__global__ __launch_bounds__(256)
void binlin_wmma_bf16_conv(const float* __restrict__ x,
                           const float* __restrict__ W,
                           const float* __restrict__ alpha,
                           const float* __restrict__ bias,
                           float* __restrict__ out)
{
  __shared__ __align__(16) __bf16 As[2][BM * LDT];
  __shared__ __align__(16) __bf16 Bs[2][BM * LDT];

  const int tid  = threadIdx.x;
  const int lane = tid & 31;
  const int wid  = tid >> 5;
  const int waveM = wid & 3;
  const int waveN = wid >> 2;

  const int m0 = blockIdx.y * BM;
  const int n0 = blockIdx.x * BN;

  int rowA[4], colT[4];
  const float* xp[4];
  const float* wp[4];
  float alw[4];
  #pragma unroll
  for (int j = 0; j < 4; ++j) {
    int idx = tid + 256 * j;
    int row = idx >> 3;
    int c4  = (idx & 7) << 2;
    rowA[j] = row;
    colT[j] = c4;
    xp[j]  = x + (size_t)(m0 + row) * K_DIM + c4;
    wp[j]  = W + (size_t)(n0 + row) * K_DIM + c4;
    alw[j] = alpha[n0 + row];
  }

  auto fill = [&](int kt, int s) {
    const int k0 = kt * BK;
    #pragma unroll
    for (int j = 0; j < 4; ++j) {
      float4 xa = *(const float4*)(xp[j] + k0);
      float4 wa = *(const float4*)(wp[j] + k0);
      float a = alw[j];
      uint2 pa, pb;
      pa.x = pack2_bf16(xa.x, xa.y);
      pa.y = pack2_bf16(xa.z, xa.w);
      pb.x = pack2_bf16(__builtin_copysignf(a, wa.x), __builtin_copysignf(a, wa.y));
      pb.y = pack2_bf16(__builtin_copysignf(a, wa.z), __builtin_copysignf(a, wa.w));
      *(uint2*)&As[s][rowA[j] * LDT + colT[j]] = pa;
      *(uint2*)&Bs[s][rowA[j] * LDT + colT[j]] = pb;
    }
  };

  v8f acc[2][4];
  const v8f vzero = {0.f, 0.f, 0.f, 0.f, 0.f, 0.f, 0.f, 0.f};
  #pragma unroll
  for (int mi = 0; mi < 2; ++mi)
    #pragma unroll
    for (int ni = 0; ni < 4; ++ni)
      acc[mi][ni] = vzero;

  fill(0, 0);
  __syncthreads();

  const int half = lane >> 4;
  const int l15  = lane & 15;

  for (int kt = 0; kt < NKT; ++kt) {
    const int s = kt & 1;

    v16bf af[2];
    #pragma unroll
    for (int mi = 0; mi < 2; ++mi) {
      int row = waveM * 32 + mi * 16 + l15;
      const char* p = (const char*)&As[s][row * LDT];
      int o = half ? 16 : 0;
      FragU f;
      f.u[0] = *(const uint4*)(p + o);
      f.u[1] = *(const uint4*)(p + o + 32);
      af[mi] = f.v;
    }
    v16bf bfr[4];
    #pragma unroll
    for (int ni = 0; ni < 4; ++ni) {
      int col = waveN * 64 + ni * 16 + l15;
      const char* p = (const char*)&Bs[s][col * LDT];
      int o = half ? 16 : 0;
      FragU f;
      f.u[0] = *(const uint4*)(p + o);
      f.u[1] = *(const uint4*)(p + o + 32);
      bfr[ni] = f.v;
    }

    if (kt + 2 < NKT) {
      __builtin_prefetch(xp[0] + (size_t)(kt + 2) * BK, 0, 0);
      __builtin_prefetch(wp[0] + (size_t)(kt + 2) * BK, 0, 0);
    }
    if (kt + 1 < NKT) fill(kt + 1, s ^ 1);

    #pragma unroll
    for (int mi = 0; mi < 2; ++mi)
      #pragma unroll
      for (int ni = 0; ni < 4; ++ni)
        acc[mi][ni] = __builtin_amdgcn_wmma_f32_16x16x32_bf16(
            false, af[mi], false, bfr[ni], (short)0, acc[mi][ni], false, false);

    __syncthreads();
  }

  #pragma unroll
  for (int mi = 0; mi < 2; ++mi) {
    int mbase = m0 + waveM * 32 + mi * 16 + half * 8;
    #pragma unroll
    for (int ni = 0; ni < 4; ++ni) {
      int n = n0 + waveN * 64 + ni * 16 + l15;
      float bv = bias[n];
      #pragma unroll
      for (int r = 0; r < 8; ++r)
        out[(size_t)(mbase + r) * N_DIM + n] = acc[mi][ni][r] + bv;
    }
  }
}

extern "C" void kernel_launch(void* const* d_in, const int* in_sizes, int n_in,
                              void* d_out, int out_size, void* d_ws, size_t ws_size,
                              hipStream_t stream) {
  (void)in_sizes; (void)n_in; (void)out_size;
  const float* x     = (const float*)d_in[0];
  const float* W     = (const float*)d_in[1];
  const float* alpha = (const float*)d_in[2];
  const float* b     = (const float*)d_in[3];
  float* out = (float*)d_out;

  const size_t xelem = (size_t)M_ROWS * K_DIM;            // 33.5M
  const size_t welem = (size_t)N_DIM * K_DIM;             // 16.8M
  const size_t need  = (xelem + welem) * sizeof(__bf16);  // ~100 MB

  dim3 grid(N_DIM / BN, M_ROWS / BM);
  dim3 block(256);

  if (ws_size >= need) {
    __bf16* xb = (__bf16*)d_ws;
    __bf16* wb = xb + xelem;
    cvt_x_bf16<<<dim3((unsigned)(xelem / (8 * 256))), block, 0, stream>>>(x, xb);
    cvt_w_bf16<<<dim3((unsigned)(welem / (8 * 256))), block, 0, stream>>>(W, alpha, wb);
    binlin_wmma_bf16_async<<<grid, block, 0, stream>>>(xb, wb, b, out);
  } else {
    binlin_wmma_bf16_conv<<<grid, block, 0, stream>>>(x, W, alpha, b, out);
  }
}
